// EquivariantGraphTransformer_50809463111706
// MI455X (gfx1250) — compile-verified, matrix operand order checked by reading, not run on hardware
//
#include <hip/hip_runtime.h>
#include <hip/hip_bf16.h>

#define H     256
#define LNUM  6
#define NHEADS 8
#define HDIM  32
#define NTOT  16384
#define NEDGE 262144
#define BG    256   // graphs
#define MAXN  64

typedef __attribute__((ext_vector_type(16))) __bf16 v16bf;
typedef __attribute__((ext_vector_type(8)))  __bf16 v8bf;
typedef __attribute__((ext_vector_type(8)))  float  v8f;

// ---------------------------------------------------------------------------
// WMMA helpers (CDNA5 wave32 fragment layouts, CDNA5 ISA §7.12.2)
// A 16x32 bf16: lane half = K-phase; per lane two contiguous 8-element chunks
// at K offsets 8*half and 16+8*half. B for X@W^T with W:[N,K] row-major loads
// with the identical pattern (lane = output column n).
// ---------------------------------------------------------------------------
__device__ __forceinline__ v16bf load_frag(const __bf16* base, int ld, int lane) {
  const int half = lane >> 4, r = lane & 15;
  const __bf16* p = base + (long)r * ld + 8 * half;
  v8bf lo = *(const v8bf*)p;
  v8bf hi = *(const v8bf*)(p + 16);
  v16bf f;
#pragma unroll
  for (int i = 0; i < 8; i++) { f[i] = lo[i]; f[8 + i] = hi[i]; }
  return f;
}

__device__ __forceinline__ v8f wmma_bf16(v16bf a, v16bf b, v8f c) {
  return __builtin_amdgcn_wmma_f32_16x16x32_bf16(false, a, false, b, (short)0, c,
                                                 false, false);
}

__device__ __forceinline__ float silu_f(float x) { return x / (1.f + __expf(-x)); }
__device__ __forceinline__ float gelu_f(float x) {
  return 0.5f * x * (1.f + erff(x * 0.70710678118654752f));
}

// Async copy: 16B per lane, global -> LDS directly (no VGPR round trip).
// Tracked by ASYNCcnt; wait with s_wait_asynccnt before the barrier.
__device__ __forceinline__ void async_g2l_b128(unsigned lds_addr, const void* gaddr) {
  asm volatile("global_load_async_to_lds_b128 %0, %1, off"
               :: "v"(lds_addr), "v"(gaddr) : "memory");
}
__device__ __forceinline__ void async_g2l_b128x2(unsigned lds_addr, const void* gaddr) {
  // INST_OFFSET applies to both LDS and global sides (CDNA5 ISA)
  asm volatile("global_load_async_to_lds_b128 %0, %1, off\n\t"
               "global_load_async_to_lds_b128 %0, %1, off offset:16"
               :: "v"(lds_addr), "v"(gaddr) : "memory");
}
__device__ __forceinline__ void wait_async0() {
  asm volatile("s_wait_asynccnt 0x0" ::: "memory");
}
__device__ __forceinline__ unsigned lds_addr_of(const void* p) {
  return (unsigned)(unsigned long long)p;  // low 32 bits = LDS offset
}

// ---------------------------------------------------------------------------
// Elementwise utility kernels
// ---------------------------------------------------------------------------
__global__ void k_cvt_bf16(const float* __restrict__ in, __bf16* __restrict__ out, long n) {
  long i = (long)blockIdx.x * blockDim.x + threadIdx.x;
  if (i < n) out[i] = (__bf16)in[i];
}

// phi_e_w1 [rows,513] -> bf16 [rows,512] (last column handled separately in fp32)
__global__ void k_repack_we1(const float* __restrict__ in, __bf16* __restrict__ out, int total) {
  int i = blockIdx.x * blockDim.x + threadIdx.x;
  if (i < total) {
    int r = i >> 9, c = i & 511;
    out[i] = (__bf16)in[(long)r * 513 + c];
  }
}

// obf = bf16(a+b); optionally of = a+b (fp32)
__global__ void k_add_cvt(const float* __restrict__ a, const float* __restrict__ b,
                          __bf16* __restrict__ obf, float* __restrict__ of, long n) {
  long i = (long)blockIdx.x * blockDim.x + threadIdx.x;
  if (i < n) {
    float s = a[i] + b[i];
    if (of) of[i] = s;
    obf[i] = (__bf16)s;
  }
}

__global__ void k_pos_update(float* __restrict__ cur, const float* __restrict__ pu,
                             const float* __restrict__ scales, int l, long n) {
  long i = (long)blockIdx.x * blockDim.x + threadIdx.x;
  if (i < n) cur[i] += scales[l] * pu[i];
}

// ---------------------------------------------------------------------------
// Generic GEMM: Y[M,N] = act(X_bf16[M,K] @ W_bf16[N,K]^T + bias), fp32 out,
// optional bf16 mirror. Block = 128 threads (4 waves), tile 64x64, K step 32.
// A staged into LDS via async-to-LDS (stride 40 = conflict-free, 16B rows);
// B frags loaded directly from global (weights are L2-resident: 192MB L2).
// ---------------------------------------------------------------------------
__global__ __launch_bounds__(128) void k_gemm(const __bf16* __restrict__ X,
                                              const __bf16* __restrict__ W,
                                              const float* __restrict__ bias,
                                              float* __restrict__ Y,
                                              __bf16* __restrict__ Ybf,
                                              int N, int K, int act) {
  __shared__ __align__(16) __bf16 At[64][40];
  const int lane = threadIdx.x & 31, wv = threadIdx.x >> 5;
  const int m0 = blockIdx.y * 64, n0 = blockIdx.x * 64;
  const int nloc = lane & 15, mb = (lane >> 4) * 8;

  v8f acc[4];
#pragma unroll
  for (int j = 0; j < 4; j++) {
    float bv = bias[n0 + j * 16 + nloc];
#pragma unroll
    for (int v = 0; v < 8; v++) acc[j][v] = bv;
  }

  const int rr = threadIdx.x >> 1;
  const int ch = (threadIdx.x & 1) * 16;
  const unsigned ldsa = lds_addr_of(&At[rr][ch]);
  for (int k0 = 0; k0 < K; k0 += 32) {
    const __bf16* src = X + (long)(m0 + rr) * K + k0 + ch;
    async_g2l_b128x2(ldsa, src);              // 32B/lane, no VGPR round trip
    __builtin_prefetch(src + 32, 0, 3);       // WGP-scope prefetch of next tile
    wait_async0();
    __syncthreads();
    v16bf a = load_frag(&At[wv * 16][0], 40, lane);
#pragma unroll
    for (int j = 0; j < 4; j++) {
      v16bf b = load_frag(W + (long)(n0 + j * 16) * K + k0, K, lane);
      acc[j] = wmma_bf16(a, b, acc[j]);
    }
    __syncthreads();
  }

#pragma unroll
  for (int j = 0; j < 4; j++)
#pragma unroll
    for (int v = 0; v < 8; v++) {
      float x = acc[j][v];
      if (act == 1) x = silu_f(x);
      else if (act == 2) x = gelu_f(x);
      long idx = (long)(m0 + wv * 16 + mb + v) * N + (n0 + j * 16 + nloc);
      Y[idx] = x;
      if (Ybf) Ybf[idx] = (__bf16)x;
    }
}

// ---------------------------------------------------------------------------
// Row LayerNorm over H=256: one wave32 per row, 8 elems/lane, shfl reductions.
// Y = LN(A [+ B]) * g + b; optional bf16 mirror. Block = 256 thr = 8 rows.
// ---------------------------------------------------------------------------
__global__ __launch_bounds__(256) void k_res_ln(const float* __restrict__ A,
                                                const float* __restrict__ Bp,
                                                const float* __restrict__ g,
                                                const float* __restrict__ b,
                                                float* __restrict__ Y,
                                                __bf16* __restrict__ Ybf) {
  const int lane = threadIdx.x & 31, wv = threadIdx.x >> 5;
  const long row = (long)blockIdx.x * 8 + wv;
  const float* pa = A + row * H + lane * 8;
  float x[8];
#pragma unroll
  for (int i = 0; i < 8; i++) x[i] = pa[i];
  if (Bp) {
    const float* pb = Bp + row * H + lane * 8;
#pragma unroll
    for (int i = 0; i < 8; i++) x[i] += pb[i];
  }
  float s = 0.f;
#pragma unroll
  for (int i = 0; i < 8; i++) s += x[i];
#pragma unroll
  for (int m = 1; m < 32; m <<= 1) s += __shfl_xor(s, m, 32);
  float mean = s * (1.f / H);
  float var = 0.f;
#pragma unroll
  for (int i = 0; i < 8; i++) { float d = x[i] - mean; var += d * d; }
#pragma unroll
  for (int m = 1; m < 32; m <<= 1) var += __shfl_xor(var, m, 32);
  float rstd = rsqrtf(var * (1.f / H) + 1e-5f);
#pragma unroll
  for (int i = 0; i < 8; i++) {
    int c = lane * 8 + i;
    float y = (x[i] - mean) * rstd * g[c] + b[c];
    Y[row * H + c] = y;
    if (Ybf) Ybf[row * H + c] = (__bf16)y;
  }
}

// ---------------------------------------------------------------------------
// Fused edge pipeline: 64 edges/block, 256 thr (8 waves).
//   m   = LN(SiLU([h_r|h_c|dist] @ We1^T + b1) @ We2^T + b2)   (kept in LDS)
//   hmsg[col] += m (global f32 atomics, L2-resident scatter)
//   coeff = tanh(SiLU(m @ Wx1^T + bx1) . wx2 + bx2)
//   posupd[col] += rel * coeff
// LN statistics combined across waves with ds_add_f32. Avoids materializing
// the 256MB/layer edge-message tensor in HBM (the roofline killer).
// Wave w: m-rows 16*(w&3), n-block 128*(w>>2)  -> 8 accum tiles per wave.
// ---------------------------------------------------------------------------
__global__ __launch_bounds__(256) void k_edge(
    const __bf16* __restrict__ hbf, const float* __restrict__ cur,
    const int* __restrict__ erow, const int* __restrict__ ecol,
    const __bf16* __restrict__ We1, const float* __restrict__ we1last,
    const float* __restrict__ be1, const __bf16* __restrict__ We2,
    const float* __restrict__ be2, const float* __restrict__ lng,
    const float* __restrict__ lnb, const __bf16* __restrict__ Wx1,
    const float* __restrict__ bx1, const float* __restrict__ wx2,
    const float* __restrict__ bx2, float* __restrict__ hmsg,
    float* __restrict__ posupd) {
  __shared__ __align__(16) __bf16 At[64][40];
  __shared__ __align__(16) __bf16 Mt[64][264];
  __shared__ int rid[64], cid[64];
  __shared__ float rel_s[64][4];
  __shared__ float rowsum[64], rowsq[64], csum[64];

  const int tid = threadIdx.x;
  const int lane = tid & 31, wv = tid >> 5;
  const int mrow0 = (wv & 3) * 16;
  const int nblk = (wv >> 2) * 128;
  const int nloc = lane & 15, mb = (lane >> 4) * 8;

  if (tid < 64) {
    int e = blockIdx.x * 64 + tid;
    int r = erow[e], c = ecol[e];
    rid[tid] = r; cid[tid] = c;
    float dx = cur[r * 3 + 0] - cur[c * 3 + 0];
    float dy = cur[r * 3 + 1] - cur[c * 3 + 1];
    float dz = cur[r * 3 + 2] - cur[c * 3 + 2];
    float ex = dx + 1e-8f, ey = dy + 1e-8f, ez = dz + 1e-8f;
    rel_s[tid][0] = dx; rel_s[tid][1] = dy; rel_s[tid][2] = dz;
    rel_s[tid][3] = sqrtf(ex * ex + ey * ey + ez * ez);
    rowsum[tid] = 0.f; rowsq[tid] = 0.f; csum[tid] = bx2[0];
  }
  __syncthreads();

  // ---- GEMM1: 64x256, K=513 (512 gathered + dist column folded into init)
  v8f acc[8];
#pragma unroll
  for (int j = 0; j < 8; j++) {
    int n = nblk + j * 16 + nloc;
    float bv = be1[n], wl = we1last[(long)n * 513];
#pragma unroll
    for (int v = 0; v < 8; v++) acc[j][v] = bv + rel_s[mrow0 + mb + v][3] * wl;
  }
  const int rr = tid >> 2;
  const int ch = (tid & 3) * 8;
  const unsigned ldsa = lds_addr_of(&At[rr][ch]);
  for (int k0 = 0; k0 < 512; k0 += 32) {
    int node = (k0 < 256) ? rid[rr] : cid[rr];
    int kk = (k0 < 256) ? k0 : (k0 - 256);
    async_g2l_b128(ldsa, hbf + (long)node * H + kk + ch);  // gathered row chunk
    wait_async0();
    __syncthreads();
    v16bf a = load_frag(&At[mrow0][0], 40, lane);
#pragma unroll
    for (int j = 0; j < 8; j++) {
      v16bf b = load_frag(We1 + (long)(nblk + j * 16) * 512 + k0, 512, lane);
      acc[j] = wmma_bf16(a, b, acc[j]);
    }
    __syncthreads();
  }
#pragma unroll
  for (int j = 0; j < 8; j++)
#pragma unroll
    for (int v = 0; v < 8; v++)
      Mt[mrow0 + mb + v][nblk + j * 16 + nloc] = (__bf16)silu_f(acc[j][v]);
  __syncthreads();

  // ---- GEMM2: 64x256, K=256 from Mt (LDS-resident)
#pragma unroll
  for (int j = 0; j < 8; j++) {
    float bv = be2[nblk + j * 16 + nloc];
#pragma unroll
    for (int v = 0; v < 8; v++) acc[j][v] = bv;
  }
  for (int k0 = 0; k0 < 256; k0 += 32) {
    v16bf a = load_frag(&Mt[mrow0][k0], 264, lane);
#pragma unroll
    for (int j = 0; j < 8; j++) {
      v16bf b = load_frag(We2 + (long)(nblk + j * 16) * H + k0, H, lane);
      acc[j] = wmma_bf16(a, b, acc[j]);
    }
  }
  // row statistics for LN (16-lane shfl halves match C-fragment layout)
  {
    float rs[8], rq[8];
#pragma unroll
    for (int v = 0; v < 8; v++) {
      rs[v] = 0.f; rq[v] = 0.f;
#pragma unroll
      for (int j = 0; j < 8; j++) { float x = acc[j][v]; rs[v] += x; rq[v] += x * x; }
#pragma unroll
      for (int m = 1; m < 16; m <<= 1) {
        rs[v] += __shfl_xor(rs[v], m, 32);
        rq[v] += __shfl_xor(rq[v], m, 32);
      }
    }
    if (nloc == 0) {
#pragma unroll
      for (int v = 0; v < 8; v++) {
        atomicAdd(&rowsum[mrow0 + mb + v], rs[v]);   // ds_add_f32
        atomicAdd(&rowsq[mrow0 + mb + v], rq[v]);
      }
    }
  }
  __syncthreads();
  // normalize, write bf16 back to Mt, scatter into hmsg with global atomics
#pragma unroll
  for (int v = 0; v < 8; v++) {
    int m = mrow0 + mb + v;
    float mean = rowsum[m] * (1.f / H);
    float var = rowsq[m] * (1.f / H) - mean * mean;
    float rstd = rsqrtf(var + 1e-5f);
    long cbase = (long)cid[m] * H;
#pragma unroll
    for (int j = 0; j < 8; j++) {
      int n = nblk + j * 16 + nloc;
      float y = (acc[j][v] - mean) * rstd * lng[n] + lnb[n];
      Mt[m][n] = (__bf16)y;
      atomicAdd(&hmsg[cbase + n], y);
    }
  }
  __syncthreads();

  // ---- GEMM3 (phi_x layer 1) + gate reduction
#pragma unroll
  for (int j = 0; j < 8; j++) {
    float bv = bx1[nblk + j * 16 + nloc];
#pragma unroll
    for (int v = 0; v < 8; v++) acc[j][v] = bv;
  }
  for (int k0 = 0; k0 < 256; k0 += 32) {
    v16bf a = load_frag(&Mt[mrow0][k0], 264, lane);
#pragma unroll
    for (int j = 0; j < 8; j++) {
      v16bf b = load_frag(Wx1 + (long)(nblk + j * 16) * H + k0, H, lane);
      acc[j] = wmma_bf16(a, b, acc[j]);
    }
  }
  {
    float pv[8];
#pragma unroll
    for (int v = 0; v < 8; v++) {
      pv[v] = 0.f;
#pragma unroll
      for (int j = 0; j < 8; j++)
        pv[v] += silu_f(acc[j][v]) * wx2[nblk + j * 16 + nloc];
#pragma unroll
      for (int m = 1; m < 16; m <<= 1) pv[v] += __shfl_xor(pv[v], m, 32);
    }
    if (nloc == 0)
#pragma unroll
      for (int v = 0; v < 8; v++) atomicAdd(&csum[mrow0 + mb + v], pv[v]);
  }
  __syncthreads();
  if (tid < 64) {
    float c = tanhf(csum[tid]);
    long pb = (long)cid[tid] * 3;
    atomicAdd(&posupd[pb + 0], rel_s[tid][0] * c);
    atomicAdd(&posupd[pb + 1], rel_s[tid][1] * c);
    atomicAdd(&posupd[pb + 2], rel_s[tid][2] * c);
  }
}

// ---------------------------------------------------------------------------
// Fused attention: one block per (graph, head). 128 thr (4 waves).
// S = (Q/sqrt(d)) K^T via WMMA (K staged [n][d] == W-style), in-register
// softmax over the 16-lane halves, O = P V via WMMA (V staged transposed).
// ---------------------------------------------------------------------------
__global__ __launch_bounds__(128) void k_attn(const float* __restrict__ qkv,
                                              __bf16* __restrict__ obf) {
  __shared__ __align__(16) __bf16 Qs[64][40];
  __shared__ __align__(16) __bf16 Ks[64][40];
  __shared__ __align__(16) __bf16 Vt[32][72];
  __shared__ __align__(16) __bf16 Ps[64][72];
  const int g = blockIdx.x, hd = blockIdx.y;
  const int lane = threadIdx.x & 31, wv = threadIdx.x >> 5;
  const int nloc = lane & 15, mb = (lane >> 4) * 8;

  {
    int r = threadIdx.x >> 1;
    int half = (threadIdx.x & 1) * 16;
    const float* base = qkv + (long)(g * 64 + r) * (3 * H);
    const float* q = base + hd * HDIM + half;
    const float* k = base + H + hd * HDIM + half;
    const float* v = base + 2 * H + hd * HDIM + half;
    const float qs = 0.17677669529663688f;  // 1/sqrt(32)
#pragma unroll
    for (int i = 0; i < 16; i++) {
      Qs[r][half + i] = (__bf16)(q[i] * qs);
      Ks[r][half + i] = (__bf16)(k[i]);
      Vt[half + i][r] = (__bf16)(v[i]);
    }
  }
  __syncthreads();

  // S = Q K^T : wave wv owns rows 16*wv, all 4 n tiles, K=32 single step
  v8f s[4];
#pragma unroll
  for (int j = 0; j < 4; j++)
#pragma unroll
    for (int v = 0; v < 8; v++) s[j][v] = 0.f;
  {
    v16bf a = load_frag(&Qs[wv * 16][0], 40, lane);
#pragma unroll
    for (int j = 0; j < 4; j++) {
      v16bf b = load_frag(&Ks[j * 16][0], 40, lane);
      s[j] = wmma_bf16(a, b, s[j]);
    }
  }
  // in-register softmax per row (row elems live in 16 lanes x 4 tiles)
#pragma unroll
  for (int v = 0; v < 8; v++) {
    float mx = -3.0e38f;
#pragma unroll
    for (int j = 0; j < 4; j++) mx = fmaxf(mx, s[j][v]);
#pragma unroll
    for (int m = 1; m < 16; m <<= 1) mx = fmaxf(mx, __shfl_xor(mx, m, 32));
    float sum = 0.f;
#pragma unroll
    for (int j = 0; j < 4; j++) { s[j][v] = __expf(s[j][v] - mx); sum += s[j][v]; }
#pragma unroll
    for (int m = 1; m < 16; m <<= 1) sum += __shfl_xor(sum, m, 32);
    float inv = 1.f / sum;
    int row = wv * 16 + mb + v;
#pragma unroll
    for (int j = 0; j < 4; j++) Ps[row][j * 16 + nloc] = (__bf16)(s[j][v] * inv);
  }
  __syncthreads();

  // O = P V : K=64 (2 steps), N=32 (2 tiles)
  v8f o[2];
#pragma unroll
  for (int j = 0; j < 2; j++)
#pragma unroll
    for (int v = 0; v < 8; v++) o[j][v] = 0.f;
#pragma unroll
  for (int k0 = 0; k0 < 64; k0 += 32) {
    v16bf a = load_frag(&Ps[wv * 16][k0], 72, lane);
#pragma unroll
    for (int j = 0; j < 2; j++) {
      v16bf b = load_frag(&Vt[j * 16][k0], 72, lane);
      o[j] = wmma_bf16(a, b, o[j]);
    }
  }
#pragma unroll
  for (int j = 0; j < 2; j++)
#pragma unroll
    for (int v = 0; v < 8; v++) {
      long idx = (long)(g * 64 + wv * 16 + mb + v) * H + hd * HDIM + j * 16 + nloc;
      obf[idx] = (__bf16)o[j][v];
    }
}

// ---------------------------------------------------------------------------
// Host orchestration
// ---------------------------------------------------------------------------
extern "C" void kernel_launch(void* const* d_in, const int* in_sizes, int n_in,
                              void* d_out, int out_size, void* d_ws, size_t ws_size,
                              hipStream_t stream) {
  (void)in_sizes; (void)n_in; (void)out_size; (void)ws_size;
  const float* x = (const float*)d_in[0];
  const float* pos = (const float*)d_in[1];
  const int* ei = (const int*)d_in[2];
  const float* pe_w1 = (const float*)d_in[5];
  const float* pe_b1 = (const float*)d_in[6];
  const float* pe_w2 = (const float*)d_in[7];
  const float* pe_b2 = (const float*)d_in[8];
  const float* pe_lg = (const float*)d_in[9];
  const float* pe_lb = (const float*)d_in[10];
  const float* ph_w1 = (const float*)d_in[11];
  const float* ph_b1 = (const float*)d_in[12];
  const float* ph_w2 = (const float*)d_in[13];
  const float* ph_b2 = (const float*)d_in[14];
  const float* ph_lg = (const float*)d_in[15];
  const float* ph_lb = (const float*)d_in[16];
  const float* px_w1 = (const float*)d_in[17];
  const float* px_b1 = (const float*)d_in[18];
  const float* px_w2 = (const float*)d_in[19];
  const float* px_b2 = (const float*)d_in[20];
  const float* ai_w = (const float*)d_in[21];
  const float* ai_b = (const float*)d_in[22];
  const float* ao_w = (const float*)d_in[23];
  const float* ao_b = (const float*)d_in[24];
  const float* lh_g = (const float*)d_in[25];
  const float* lh_b = (const float*)d_in[26];
  const float* la_g = (const float*)d_in[27];
  const float* la_b = (const float*)d_in[28];
  const float* f_w1 = (const float*)d_in[29];
  const float* f_b1 = (const float*)d_in[30];
  const float* f_w2 = (const float*)d_in[31];
  const float* f_b2 = (const float*)d_in[32];
  const float* pscal = (const float*)d_in[33];

  float* h = (float*)d_out;                 // [NTOT,H]  (final output slot 0)
  float* cur = h + (size_t)NTOT * H;        // [NTOT,3]  (final output slot 1)

  char* wp = (char*)d_ws;
  auto alloc = [&](size_t bytes) -> void* {
    void* p = (void*)wp;
    wp += (bytes + 255) & ~(size_t)255;
    return p;
  };
  __bf16* we1 = (__bf16*)alloc((size_t)LNUM * 256 * 512 * 2);
  __bf16* we2 = (__bf16*)alloc((size_t)LNUM * 256 * 256 * 2);
  __bf16* wh1 = (__bf16*)alloc((size_t)LNUM * 256 * 256 * 2);
  __bf16* wh2 = (__bf16*)alloc((size_t)LNUM * 256 * 256 * 2);
  __bf16* wx1 = (__bf16*)alloc((size_t)LNUM * 256 * 256 * 2);
  __bf16* wai = (__bf16*)alloc((size_t)LNUM * 768 * 256 * 2);
  __bf16* wao = (__bf16*)alloc((size_t)LNUM * 256 * 256 * 2);
  __bf16* wf1 = (__bf16*)alloc((size_t)LNUM * 1024 * 256 * 2);
  __bf16* wf2 = (__bf16*)alloc((size_t)LNUM * 256 * 1024 * 2);
  __bf16* hbf = (__bf16*)alloc((size_t)NTOT * H * 2);
  __bf16* bufbf = (__bf16*)alloc((size_t)NTOT * H * 2);       // huin / obf
  __bf16* bigbf = (__bf16*)alloc((size_t)NTOT * 1024 * 2);    // t1 / ff1 bf16
  float* hmsg = (float*)alloc((size_t)NTOT * H * 4);
  float* posupd = (float*)alloc((size_t)NTOT * 3 * 4);
  float* tmpf = (float*)alloc((size_t)NTOT * 1024 * 4);       // qkv / ff1 / hu
  float* tmp2f = (float*)alloc((size_t)NTOT * H * 4);

  hipMemcpyAsync(h, x, (size_t)NTOT * H * 4, hipMemcpyDeviceToDevice, stream);
  hipMemcpyAsync(cur, pos, (size_t)NTOT * 3 * 4, hipMemcpyDeviceToDevice, stream);
  {
    long n = (long)NTOT * H;
    k_cvt_bf16<<<(n + 255) / 256, 256, 0, stream>>>(x, hbf, n);
  }
  k_repack_we1<<<(LNUM * 256 * 512 + 255) / 256, 256, 0, stream>>>(pe_w1, we1,
                                                                   LNUM * 256 * 512);
  auto cvt = [&](const float* src, __bf16* dst, long n) {
    k_cvt_bf16<<<(n + 255) / 256, 256, 0, stream>>>(src, dst, n);
  };
  cvt(pe_w2, we2, (long)LNUM * 256 * 256);
  cvt(ph_w1, wh1, (long)LNUM * 256 * 256);
  cvt(ph_w2, wh2, (long)LNUM * 256 * 256);
  cvt(px_w1, wx1, (long)LNUM * 256 * 256);
  cvt(ai_w, wai, (long)LNUM * 768 * 256);
  cvt(ao_w, wao, (long)LNUM * 256 * 256);
  cvt(f_w1, wf1, (long)LNUM * 1024 * 256);
  cvt(f_w2, wf2, (long)LNUM * 256 * 1024);

  const int* erow = ei;
  const int* ecol = ei + NEDGE;

  for (int l = 0; l < LNUM; l++) {
    hipMemsetAsync(hmsg, 0, (size_t)NTOT * H * 4, stream);
    hipMemsetAsync(posupd, 0, (size_t)NTOT * 3 * 4, stream);
    // fused EDM edge pipeline
    k_edge<<<NEDGE / 64, 256, 0, stream>>>(
        hbf, cur, erow, ecol, we1 + (size_t)l * 256 * 512,
        pe_w1 + (size_t)l * 256 * 513 + 512, pe_b1 + l * 256,
        we2 + (size_t)l * 256 * 256, pe_b2 + l * 256, pe_lg + l * 256,
        pe_lb + l * 256, wx1 + (size_t)l * 256 * 256, px_b1 + l * 256,
        px_w2 + l * 256, px_b2 + l, hmsg, posupd);
    // hu_in = h + h_msg (bf16)
    {
      long n = (long)NTOT * H;
      k_add_cvt<<<(n + 255) / 256, 256, 0, stream>>>(h, hmsg, bufbf, nullptr, n);
    }
    // phi_h MLP + LN
    k_gemm<<<dim3(H / 64, NTOT / 64), 128, 0, stream>>>(
        bufbf, wh1 + (size_t)l * 256 * 256, ph_b1 + l * 256, tmpf, bigbf, H, H, 1);
    k_gemm<<<dim3(H / 64, NTOT / 64), 128, 0, stream>>>(
        bigbf, wh2 + (size_t)l * 256 * 256, ph_b2 + l * 256, tmp2f, nullptr, H, H, 0);
    k_res_ln<<<NTOT / 8, 256, 0, stream>>>(tmp2f, nullptr, ph_lg + l * 256,
                                           ph_lb + l * 256, tmpf, nullptr);
    // h = LN(h + hu)
    k_res_ln<<<NTOT / 8, 256, 0, stream>>>(h, tmpf, lh_g + l * 256, lh_b + l * 256,
                                           h, hbf);
    // cur += scale * pos_upd
    {
      long n = (long)NTOT * 3;
      k_pos_update<<<(n + 255) / 256, 256, 0, stream>>>(cur, posupd, pscal, l, n);
    }
    // attention: qkv projection, fused attention, output projection, LN
    k_gemm<<<dim3(768 / 64, NTOT / 64), 128, 0, stream>>>(
        hbf, wai + (size_t)l * 768 * 256, ai_b + l * 768, tmpf, nullptr, 768, H, 0);
    k_attn<<<dim3(BG, NHEADS), 128, 0, stream>>>(tmpf, bufbf);
    k_gemm<<<dim3(H / 64, NTOT / 64), 128, 0, stream>>>(
        bufbf, wao + (size_t)l * 256 * 256, ao_b + l * 256, tmp2f, nullptr, H, H, 0);
    k_res_ln<<<NTOT / 8, 256, 0, stream>>>(h, tmp2f, la_g + l * 256, la_b + l * 256,
                                           h, hbf);
    // feed-forward + residual
    k_gemm<<<dim3(1024 / 64, NTOT / 64), 128, 0, stream>>>(
        hbf, wf1 + (size_t)l * 1024 * 256, f_b1 + l * 1024, tmpf, bigbf, 1024, H, 2);
    k_gemm<<<dim3(H / 64, NTOT / 64), 128, 0, stream>>>(
        bigbf, wf2 + (size_t)l * 256 * 1024, f_b2 + l * 256, tmp2f, nullptr, H, 1024, 0);
    {
      long n = (long)NTOT * H;
      k_add_cvt<<<(n + 255) / 256, 256, 0, stream>>>(h, tmp2f, hbf, h, n);
    }
  }
}
